// GCN_5282809775007
// MI455X (gfx1250) — matrix operand order repaired and checked by reading, not compiled, not run on hardware
//
#include <hip/hip_runtime.h>
#include <hip/hip_bf16.h>

// ---------------------------------------------------------------------------
// GCN (3-layer) for MI455X / gfx1250 (wave32, WMMA).
//   layer: H = X @ W (WMMA f16->f32), Agg[dst] += H[src] (atomic f32, L2-resident),
//          BN(train-mode, biased var) + ReLU, final log_softmax over D=128.
// ---------------------------------------------------------------------------

#define DFEAT 128
#define NLAYERS 3
#define BN_EPS 1e-5f

typedef __attribute__((ext_vector_type(16))) _Float16 v16h;
typedef __attribute__((ext_vector_type(8)))  _Float16 v8h;
typedef __attribute__((ext_vector_type(8)))  float    v8f;

#define HCAT(lo, hi) __builtin_shufflevector(lo, hi, 0, 1, 2, 3, 4, 5, 6, 7, \
                                             8, 9, 10, 11, 12, 13, 14, 15)
#define WMMA_F16(a, b, c) __builtin_amdgcn_wmma_f32_16x16x32_f16( \
        false, (a), false, (b), (short)0, (c), false, false)

// ---------------------------------------------------------------------------
// Pre-pack W (f32 row-major [layer][k][n]) into f16 B-fragment order:
//   frag(layer, ntile, kchunk): 32 lanes x 16 halves contiguous.
//   B layout (32x16 f16): lanes 0-15 -> K 0..15 (2 K per dword), lanes 16-31 -> K 16..31.
// One thread per dword (= half2). Total dwords: 3 * 8 * 4 * 32 * 8 = 24576.
// ---------------------------------------------------------------------------
__global__ void gcn_prep_w(const float* __restrict__ Ws, _Float16* __restrict__ Wfrag) {
    int t = blockIdx.x * blockDim.x + threadIdx.x;
    if (t >= NLAYERS * 8 * 4 * 32 * 8) return;
    int j     = t & 7;          // dword index within lane (8 dwords = 16 halves)
    int lane  = (t >> 3) & 31;
    int kc    = (t >> 8) & 3;   // k-chunk of 32
    int nt    = (t >> 10) & 7;  // n-tile of 16
    int layer = t >> 13;
    int n = nt * 16 + (lane & 15);
    int k = kc * 32 + ((lane < 16) ? 0 : 16) + 2 * j;
    const float* W = Ws + layer * DFEAT * DFEAT;
    Wfrag[2 * t + 0] = (_Float16)W[(k + 0) * DFEAT + n];
    Wfrag[2 * t + 1] = (_Float16)W[(k + 1) * DFEAT + n];
}

// ---------------------------------------------------------------------------
// GEMM: H = X @ W via v_wmma_f32_16x16x32_f16.
// Block = 256 threads = 8 waves; wave w owns n-tile w (16 cols); K=128 ->
// 4 chained WMMAs. The 16x128 X tile is converted f32->f16 ONCE while staging
// into LDS (4 KB/tile), double-buffered so the next tile's global loads overlap
// the current tile's WMMA chain. All 8 A-fragment ds_load_b128 are issued
// before the first WMMA so dscnt waits stagger across the chain.
// ---------------------------------------------------------------------------

// stage one 16x128 tile as f16 into LDS; rows >= nRows are zero-padded
__device__ __forceinline__ void gcn_stage_tile(_Float16* __restrict__ dst,
                                               const float* __restrict__ X,
                                               int mBase, int nRows, int tid) {
    int row = tid >> 4;            // 16 threads per row
    int col = (tid & 15) * 8;      // 8 consecutive floats per thread
    float4 f0 = make_float4(0.f, 0.f, 0.f, 0.f);
    float4 f1 = f0;
    if (mBase + row < nRows) {
        const float4* p = (const float4*)(X + (long long)(mBase + row) * DFEAT + col);
        f0 = p[0];
        f1 = p[1];
    }
    v8h h;
    h[0] = (_Float16)f0.x; h[1] = (_Float16)f0.y;
    h[2] = (_Float16)f0.z; h[3] = (_Float16)f0.w;
    h[4] = (_Float16)f1.x; h[5] = (_Float16)f1.y;
    h[6] = (_Float16)f1.z; h[7] = (_Float16)f1.w;
    ((v8h*)dst)[tid] = h;          // one 16B ds_store_b128, row-major f16 tile
}

__global__ void __launch_bounds__(256)
gcn_gemm_wmma(const float* __restrict__ X, const _Float16* __restrict__ Wfrag,
              float* __restrict__ H, int nRows, int nTiles) {
    __shared__ _Float16 tile[2][16 * DFEAT];

    const int tid  = threadIdx.x;
    const int wave = tid >> 5;
    const int lane = tid & 31;

    // B fragments for this wave's n-tile: 4 k-chunks, contiguous 32B per lane
    const v16h* wf = (const v16h*)Wfrag + (wave * 4) * 32 + lane;
    const v16h b0 = wf[0 * 32];
    const v16h b1 = wf[1 * 32];
    const v16h b2 = wf[2 * 32];
    const v16h b3 = wf[3 * 32];

    // A-fragment addressing (contiguous k order, same as the B packing):
    // lane<16:  row M=lane,    K = kc*32 + {0..15}
    // lane>=16: row M=lane-16, K = kc*32 + {16..31}
    const int  mrow = lane & 15;
    const int  hsel = (lane < 16) ? 0 : 2;     // v8h-granule select (koff/8)
    const int  nCol = wave * 16 + (lane & 15); // C store column
    const int  mOff = (lane < 16) ? 0 : 8;     // C store row offset

    int t = blockIdx.x;
    if (t >= nTiles) return;
    gcn_stage_tile(tile[0], X, t * 16, nRows, tid);
    int p = 0;

    for (; t < nTiles; t += gridDim.x, p ^= 1) {
        __syncthreads();  // tile[p] staged by all threads

        int tn = t + gridDim.x;
        if (tn < nTiles) gcn_stage_tile(tile[p ^ 1], X, tn * 16, nRows, tid);

        // preload all 8 A-fragment halves so dscnt waits stagger across WMMAs
        const v8h* ar = (const v8h*)(tile[p] + mrow * DFEAT);
        v8h lo0 = ar[0 * 4 + hsel], hi0 = ar[0 * 4 + 1 + hsel];
        v8h lo1 = ar[1 * 4 + hsel], hi1 = ar[1 * 4 + 1 + hsel];
        v8h lo2 = ar[2 * 4 + hsel], hi2 = ar[2 * 4 + 1 + hsel];
        v8h lo3 = ar[3 * 4 + hsel], hi3 = ar[3 * 4 + 1 + hsel];

        v8f c = {};
        c = WMMA_F16(HCAT(lo0, hi0), b0, c);
        c = WMMA_F16(HCAT(lo1, hi1), b1, c);
        c = WMMA_F16(HCAT(lo2, hi2), b2, c);
        c = WMMA_F16(HCAT(lo3, hi3), b3, c);

        // C layout: lane&15 -> N col; vgpr r -> M = r + (lane<16 ? 0 : 8)
        const int mBase = t * 16;
        if (mBase + 16 <= nRows) {
            // full tile: one address computation, 8 stores at immediate offsets
            float* hp = H + (long long)(mBase + mOff) * DFEAT + nCol;
            #pragma unroll
            for (int r = 0; r < 8; ++r) hp[r * DFEAT] = c[r];
        } else {
            #pragma unroll
            for (int r = 0; r < 8; ++r) {
                int m = mBase + mOff + r;
                if (m < nRows) H[(long long)m * DFEAT + nCol] = c[r];
            }
        }
    }
}

// ---------------------------------------------------------------------------
// Agg[dst[e], :] += H[src[e], :]   (one wave per edge; lane handles 4 floats)
// H (51 MB) is L2-resident on the 192 MB global L2, so gather+atomics are
// L2-bound, not HBM-bound.
// ---------------------------------------------------------------------------
__global__ void __launch_bounds__(256)
gcn_scatter_add(const float* __restrict__ H, const int* __restrict__ src,
                const int* __restrict__ dst, float* __restrict__ Agg,
                long long nEdges) {
    long long tid = (long long)blockIdx.x * blockDim.x + threadIdx.x;
    int q = (int)(tid & 31);
    long long e = tid >> 5;
    long long estride = ((long long)gridDim.x * blockDim.x) >> 5;
    for (; e < nEdges; e += estride) {
        int s = src[e];
        int d = dst[e];
        float4 v = ((const float4*)(H + (long long)s * DFEAT))[q];
        float* out = Agg + (long long)d * DFEAT + 4 * q;
        atomicAdd(out + 0, v.x);
        atomicAdd(out + 1, v.y);
        atomicAdd(out + 2, v.z);
        atomicAdd(out + 3, v.w);
    }
}

// ---------------------------------------------------------------------------
// Per-channel sum / sum-of-squares with LDS pre-reduction, atomics into stats.
// stats[0..127] = sum, stats[128..255] = sumsq. Block = 256 = 2 rows/iter.
// ---------------------------------------------------------------------------
__global__ void __launch_bounds__(256)
gcn_bn_stats(const float* __restrict__ X, float* __restrict__ stats, int nRows) {
    __shared__ float sh[256];
    int c = threadIdx.x & 127;
    int rb = threadIdx.x >> 7;  // 0 or 1
    float s = 0.f, s2 = 0.f;
    for (int r = blockIdx.x * 2 + rb; r < nRows; r += gridDim.x * 2) {
        float v = X[(long long)r * DFEAT + c];
        s += v;
        s2 += v * v;
    }
    sh[threadIdx.x] = s;
    __syncthreads();
    if (threadIdx.x < 128) atomicAdd(&stats[threadIdx.x], sh[threadIdx.x] + sh[threadIdx.x + 128]);
    __syncthreads();
    sh[threadIdx.x] = s2;
    __syncthreads();
    if (threadIdx.x < 128) atomicAdd(&stats[128 + threadIdx.x], sh[threadIdx.x] + sh[threadIdx.x + 128]);
}

// ---------------------------------------------------------------------------
// In-place BN(train, biased var) + ReLU.
// ---------------------------------------------------------------------------
__global__ void __launch_bounds__(256)
gcn_bn_relu(float* __restrict__ X, const float* __restrict__ stats,
            const float* __restrict__ gamma, const float* __restrict__ beta,
            int nRows, float invN) {
    long long total = (long long)nRows * DFEAT;
    long long stride = (long long)gridDim.x * blockDim.x;
    for (long long i = (long long)blockIdx.x * blockDim.x + threadIdx.x; i < total; i += stride) {
        int c = (int)(i & (DFEAT - 1));
        float mu = stats[c] * invN;
        float var = stats[128 + c] * invN - mu * mu;
        float v = (X[i] - mu) * rsqrtf(var + BN_EPS) * gamma[c] + beta[c];
        X[i] = v > 0.f ? v : 0.f;
    }
}

// ---------------------------------------------------------------------------
// log_softmax over D=128: one wave per row, shfl_xor reductions (wave32).
// ---------------------------------------------------------------------------
__global__ void __launch_bounds__(256)
gcn_log_softmax(const float* __restrict__ X, float* __restrict__ Out, int nRows) {
    int lane = threadIdx.x & 31;
    int wpb = blockDim.x >> 5;
    long long r = (long long)blockIdx.x * wpb + (threadIdx.x >> 5);
    long long rstride = (long long)gridDim.x * wpb;
    for (; r < nRows; r += rstride) {
        float4 v = ((const float4*)(X + r * DFEAT))[lane];
        float m = fmaxf(fmaxf(v.x, v.y), fmaxf(v.z, v.w));
        #pragma unroll
        for (int o = 16; o > 0; o >>= 1) m = fmaxf(m, __shfl_xor(m, o, 32));
        float s = expf(v.x - m) + expf(v.y - m) + expf(v.z - m) + expf(v.w - m);
        #pragma unroll
        for (int o = 16; o > 0; o >>= 1) s += __shfl_xor(s, o, 32);
        float lse = m + logf(s);
        float4 o4 = make_float4(v.x - lse, v.y - lse, v.z - lse, v.w - lse);
        ((float4*)(Out + r * DFEAT))[lane] = o4;
    }
}

// ---------------------------------------------------------------------------
// Orchestration. d_in order: x[N*128], edge_index[2*E], Ws[3*128*128],
// gammas[2*128], betas[2*128]. d_out: N*128 f32.
// ---------------------------------------------------------------------------
extern "C" void kernel_launch(void* const* d_in, const int* in_sizes, int n_in,
                              void* d_out, int out_size, void* d_ws, size_t ws_size,
                              hipStream_t stream) {
    const float* x      = (const float*)d_in[0];
    const int*   eidx   = (const int*)d_in[1];
    const float* Ws     = (const float*)d_in[2];
    const float* gammas = (const float*)d_in[3];
    const float* betas  = (const float*)d_in[4];

    const int       N = in_sizes[0] / DFEAT;
    const long long E = (long long)in_sizes[1] / 2;
    const int* src = eidx;
    const int* dst = eidx + E;

    // workspace layout: two N*128 f32 ping-pong buffers, 256-float stats, W frags
    float*    bufA  = (float*)d_ws;                   // GEMM output H
    float*    bufB  = bufA + (long long)N * DFEAT;    // aggregation / next-layer X
    float*    stats = bufB + (long long)N * DFEAT;    // 256 floats
    _Float16* wfrag = (_Float16*)(stats + 256);       // 3*16384 halves

    const size_t nodeBytes = (size_t)N * DFEAT * sizeof(float);
    const int mTiles = (N + 15) / 16;
    const int gemmBlocks = (mTiles < 4096) ? mTiles : 4096;

    // pack weights into f16 fragment order (24576 dwords)
    gcn_prep_w<<<(NLAYERS * 8 * 4 * 32 * 8 + 255) / 256, 256, 0, stream>>>(Ws, wfrag);

    const int scatterBlocks = 65536;  // grid-stride: E*32 / 256 lanes of work

    for (int layer = 0; layer < NLAYERS; ++layer) {
        const float* xin = (layer == 0) ? x : bufB;

        // H = X @ W[layer]
        gcn_gemm_wmma<<<gemmBlocks, 256, 0, stream>>>(xin, wfrag + layer * 16384,
                                                      bufA, N, mTiles);

        // Agg = scatter_add(H[src] -> dst)
        hipMemsetAsync(bufB, 0, nodeBytes, stream);
        gcn_scatter_add<<<scatterBlocks, 256, 0, stream>>>(bufA, src, dst, bufB, E);

        if (layer != NLAYERS - 1) {
            hipMemsetAsync(stats, 0, 256 * sizeof(float), stream);
            gcn_bn_stats<<<1024, 256, 0, stream>>>(bufB, stats, N);
            gcn_bn_relu<<<4096, 256, 0, stream>>>(bufB, stats,
                                                  gammas + layer * DFEAT,
                                                  betas + layer * DFEAT,
                                                  N, 1.0f / (float)N);
        }
    }

    // log_softmax(bufB) -> d_out   (8 rows per block)
    gcn_log_softmax<<<(N + 7) / 8, 256, 0, stream>>>(bufB, (float*)d_out, N);
}
